// GRNN_6777458393745
// MI455X (gfx1250) — matrix operand order repaired and checked by reading, not compiled, not run on hardware
//
#include <hip/hip_runtime.h>
#include <hip/hip_bf16.h>

#define BATCHN 16
#define NN 1024
#define TT 64
#define PP 8
#define QQ 8
#define HH 64

typedef __attribute__((ext_vector_type(16))) _Float16 v16h;
typedef __attribute__((ext_vector_type(8)))  _Float16 v8h;
typedef __attribute__((ext_vector_type(8)))  float    v8f;

__device__ __forceinline__ v16h cat8(v8h lo, v8h hi) {
    return __builtin_shufflevector(lo, hi, 0,1,2,3,4,5,6,7,8,9,10,11,12,13,14,15);
}

__device__ __forceinline__ float fast_tanh(float x) {
    // tanh(x) = (e^{2x}-1)/(e^{2x}+1), exp -> v_exp_f32, div -> v_rcp_f32
    float e = __expf(2.0f * x);
    return (e - 1.0f) * __builtin_amdgcn_rcpf(e + 1.0f);
}

// ---------------------------------------------------------------------------
// Prep: S -> f16, W -> W^T f16, zero Zt buffer 0, x_traj[:,0] = x0
// ---------------------------------------------------------------------------
__global__ __launch_bounds__(256)
void grnn_prep(const float* __restrict__ S, const float* __restrict__ W,
               const float* __restrict__ x0,
               _Float16* __restrict__ S16, _Float16* __restrict__ Wt16,
               _Float16* __restrict__ Zt0, float* __restrict__ x_traj)
{
    const size_t i = (size_t)blockIdx.x * blockDim.x + threadIdx.x;   // 0 .. 1M-1
    if (i < (size_t)NN * NN) {
        S16[i] = (_Float16)S[i];
        Zt0[i] = (_Float16)0.0f;      // BATCHN*HH*NN == NN*NN == 1M elements
    }
    if (i < (size_t)HH * HH) {
        const int k = (int)(i >> 6);
        const int h = (int)(i & 63);
        Wt16[h * HH + k] = (_Float16)W[i];   // Wt[h][k] = W[k][h]
    }
    if (i < (size_t)BATCHN * NN * PP) {
        const size_t b   = i / ((size_t)NN * PP);
        const size_t rem = i % ((size_t)NN * PP);
        x_traj[b * (size_t)(TT + 1) * NN * PP + rem] = x0[i];
    }
}

// ---------------------------------------------------------------------------
// One recurrence step. Grid: (NN/128, BATCHN), block 256 (8 waves).
// Wave w owns rows [i0, i0+16) of one batch, full H=64 columns.
// ---------------------------------------------------------------------------
__global__ __launch_bounds__(256)
void grnn_step(const _Float16* __restrict__ S16,
               const _Float16* __restrict__ Wt16,
               const _Float16* __restrict__ ZtR,   // Z_t   (read),  [b][h][j]
               _Float16* __restrict__ ZtW,         // Z_t+1 (write), [b][h][j]
               const float* __restrict__ Amat,     // (P,H)
               const float* __restrict__ Bmat,     // (H,Q)
               float* __restrict__ x_traj,         // (B,T+1,N,P)
               float* __restrict__ u_traj,         // (B,T,N,Q)
               int t)
{
    __shared__ float    A_lds[PP * HH];
    __shared__ float    B_lds[HH * QQ];
    __shared__ float    x_lds[8][16 * PP];
    __shared__ _Float16 stage [8][16 * HH];   // SZ tile (A-layout source)
    __shared__ _Float16 zstage[8][16 * HH];   // Z_new tile

    const int b    = blockIdx.y;
    const int tid  = threadIdx.x;
    const int wave = tid >> 5;
    const int lane = tid & 31;
    const int lrow = lane & 15;
    const int half = lane >> 4;
    const int i0   = blockIdx.x * 128 + wave * 16;

    for (int i = tid; i < PP * HH; i += 256) A_lds[i] = Amat[i];
    for (int i = tid; i < HH * QQ; i += 256) B_lds[i] = Bmat[i];
    const float* xcur = x_traj + (((size_t)b * (TT + 1) + t) * NN + i0) * PP;
    for (int i = lane; i < 16 * PP; i += 32) x_lds[wave][i] = xcur[i];
    __syncthreads();

    // ---------------- Phase 1: SZ = S @ Z  (16 x 64 tile) ----------------
    v8f acc[4];
    #pragma unroll
    for (int hb = 0; hb < 4; ++hb)
        acc[hb] = (v8f){0.f,0.f,0.f,0.f,0.f,0.f,0.f,0.f};

    const _Float16* Srow = S16 + (size_t)(i0 + lrow) * NN;
    const _Float16* Ztb  = ZtR + (size_t)b * HH * NN;

    #pragma unroll 2
    for (int k0 = 0; k0 < NN; k0 += 32) {
        // A-layout: e=0..15 -> K=k0+16*half+e
        v8h alo = *(const v8h*)(Srow + k0 + 16 * half);
        v8h ahi = *(const v8h*)(Srow + k0 + 16 * half + 8);
        v16h a  = cat8(alo, ahi);
        #pragma unroll
        for (int hb = 0; hb < 4; ++hb) {
            // B-layout: col = lrow, K = e + 16*half -> contiguous in Zt[b][h][j]
            const v16h bm = *(const v16h*)(Ztb + (size_t)(hb * 16 + lrow) * NN
                                               + k0 + 16 * half);
            acc[hb] = __builtin_amdgcn_wmma_f32_16x16x32_f16(
                false, a, false, bm, (short)0, acc[hb], false, false);
        }
    }

    // ------- Phase 2: Z_new = tanh(SZ @ W + x @ A) -------
    // stage SZ (C-layout -> row-major f16 in LDS)
    #pragma unroll
    for (int hb = 0; hb < 4; ++hb)
        #pragma unroll
        for (int r = 0; r < 8; ++r)
            stage[wave][(r + 8 * half) * HH + hb * 16 + lrow] = (_Float16)acc[hb][r];
    __builtin_amdgcn_wave_barrier();   // same-wave DS ops are in-order; pin compiler

    // C initializer: x @ A (K = 8, plain FMAs)
    v8f pre[4];
    #pragma unroll
    for (int hb = 0; hb < 4; ++hb) {
        #pragma unroll
        for (int r = 0; r < 8; ++r) {
            const int row = r + 8 * half, col = hb * 16 + lrow;
            float c = 0.f;
            #pragma unroll
            for (int p = 0; p < PP; ++p)
                c += x_lds[wave][row * PP + p] * A_lds[p * HH + col];
            pre[hb][r] = c;
        }
    }

    // pre += SZ @ W   (K = 64 -> two WMMA chunks per output tile)
    #pragma unroll
    for (int k0 = 0; k0 < HH; k0 += 32) {
        v8h alo = *(const v8h*)(&stage[wave][lrow * HH + k0 + 16 * half]);
        v8h ahi = *(const v8h*)(&stage[wave][lrow * HH + k0 + 16 * half + 8]);
        v16h a  = cat8(alo, ahi);
        #pragma unroll
        for (int hb = 0; hb < 4; ++hb) {
            const v16h bm = *(const v16h*)(Wt16 + (size_t)(hb * 16 + lrow) * HH
                                                + k0 + 16 * half);
            pre[hb] = __builtin_amdgcn_wmma_f32_16x16x32_f16(
                false, a, false, bm, (short)0, pre[hb], false, false);
        }
    }

    // tanh; stash Z_new in LDS and write transposed to Zt_{t+1}
    _Float16* ZtbW = ZtW + (size_t)b * HH * NN;
    #pragma unroll
    for (int hb = 0; hb < 4; ++hb)
        #pragma unroll
        for (int r = 0; r < 8; ++r) {
            const int row = r + 8 * half, col = hb * 16 + lrow;
            const _Float16 zh = (_Float16)fast_tanh(pre[hb][r]);
            zstage[wave][row * HH + col] = zh;
            ZtbW[(size_t)col * NN + (i0 + row)] = zh;
        }
    __builtin_amdgcn_wave_barrier();

    // ------- Phase 3: u = Z_new @ B ; x_{t+1} = x_t + u -------
    // lane handles row = lrow, q = half*4 + qi
    float uq[4];
    #pragma unroll
    for (int qi = 0; qi < 4; ++qi) {
        const int q = half * 4 + qi;
        float s = 0.f;
        #pragma unroll 8
        for (int h = 0; h < HH; ++h)
            s += (float)zstage[wave][lrow * HH + h] * B_lds[h * QQ + q];
        uq[qi] = s;
    }
    const int grow = i0 + lrow;
    float* xnext = x_traj + (((size_t)b * (TT + 1) + (t + 1)) * NN + grow) * PP;
    float* urow  = u_traj + (((size_t)b * TT + t) * NN + grow) * QQ;
    #pragma unroll
    for (int qi = 0; qi < 4; ++qi) {
        const int q = half * 4 + qi;
        urow[q]  = uq[qi];
        xnext[q] = x_lds[wave][lrow * PP + q] + uq[qi];
    }
}

// ---------------------------------------------------------------------------
extern "C" void kernel_launch(void* const* d_in, const int* in_sizes, int n_in,
                              void* d_out, int out_size, void* d_ws, size_t ws_size,
                              hipStream_t stream)
{
    const float* x0 = (const float*)d_in[0];
    const float* S  = (const float*)d_in[1];
    const float* A  = (const float*)d_in[2];
    const float* Bm = (const float*)d_in[3];
    const float* W  = (const float*)d_in[4];

    float* out    = (float*)d_out;
    float* x_traj = out;                                        // (B,T+1,N,P)
    float* u_traj = out + (size_t)BATCHN * (TT + 1) * NN * PP;  // (B,T,N,Q)

    // Workspace: S16 | Zt[0] | Zt[1] | Wt16   (all f16)
    _Float16* S16  = (_Float16*)d_ws;
    _Float16* Zt0  = S16 + (size_t)NN * NN;
    _Float16* Zt1  = Zt0 + (size_t)BATCHN * HH * NN;
    _Float16* Wt16 = Zt1 + (size_t)BATCHN * HH * NN;

    grnn_prep<<<(NN * NN) / 256, 256, 0, stream>>>(S, W, x0, S16, Wt16, Zt0, x_traj);

    dim3 grid(NN / 128, BATCHN);
    for (int t = 0; t < TT; ++t) {
        _Float16* zr = (t & 1) ? Zt1 : Zt0;
        _Float16* zw = (t & 1) ? Zt0 : Zt1;
        grnn_step<<<grid, 256, 0, stream>>>(S16, Wt16, zr, zw, A, Bm,
                                            x_traj, u_traj, t);
    }
}
